// MultiHeadAttention_55860344652116
// MI455X (gfx1250) — matrix operand order repaired and checked by reading, not compiled, not run on hardware
//
#include <hip/hip_runtime.h>
#include <hip/hip_bf16.h>
#include <math.h>

#define D_MODEL 1024
#define N_HEADS 16
#define HEAD_DIM 64
#define BSZ 4
#define SEQ 2048
#define MROWS (BSZ * SEQ) // 8192

typedef __attribute__((ext_vector_type(16))) __bf16 bf16x16;
typedef __attribute__((ext_vector_type(8)))  __bf16 bf16x8;
typedef __attribute__((ext_vector_type(8)))  float  f32x8;
typedef __attribute__((ext_vector_type(4)))  int    i32x4;

__device__ inline f32x8 zero_f32x8() {
  f32x8 z;
#pragma unroll
  for (int i = 0; i < 8; ++i) z[i] = 0.0f;
  return z;
}

// Low 32 bits of a generic pointer to a __shared__ object == LDS byte offset
// (flat LDS aperture mapping).
__device__ inline uint32_t lds_off(const void* p) {
  return (uint32_t)(uintptr_t)p;
}

// CDNA5 async global->LDS copy, 16 bytes per lane. Tracked by ASYNCcnt.
__device__ inline void async_copy_b128(uint32_t lds_addr, const void* gptr) {
  asm volatile("global_load_async_to_lds_b128 %0, %1, off"
               :: "v"(lds_addr), "v"((unsigned long long)(uintptr_t)gptr)
               : "memory");
}
__device__ inline void wait_async0() {
  asm volatile("s_wait_asynccnt 0" ::: "memory");
}

// A-operand (16-bit A 16x32): lane = 16*khalf + m;
// slots s <- K = kbase + khalf*16 + s.
__device__ inline bf16x16 wmma_a_frag(const __bf16* rowptr, int kbase, int lane) {
  return *(const bf16x16*)(rowptr + kbase + ((lane >> 4) * 16));
}

// B-operand from a "B^T row-major" buffer (row n holds K contiguously):
// lane = 16*khalf + n; slots s <- K = kbase + khalf*16 + s.
__device__ inline bf16x16 wmma_b_frag(const __bf16* nrowptr, int kbase, int lane) {
  return *(const bf16x16*)(nrowptr + kbase + ((lane >> 4) * 16));
}

__device__ inline f32x8 wmma_bf16(bf16x16 a, bf16x16 b, f32x8 c) {
  return __builtin_amdgcn_wmma_f32_16x16x32_bf16(
      false, a, false, b, (short)0, c, false, false);
}

// ---------------------------------------------------------------------------
// fp32 -> bf16 bulk conversion (one-time; removes cvt from all hot loops).
// ---------------------------------------------------------------------------
__global__ __launch_bounds__(256) void cvt_f32_bf16_kernel(
    const float* __restrict__ x, __bf16* __restrict__ y) {
  const size_t i = ((size_t)blockIdx.x * 256 + threadIdx.x) * 8;
  float4 a = *(const float4*)(x + i);
  float4 b = *(const float4*)(x + i + 4);
  bf16x8 o;
  o[0] = (__bf16)a.x; o[1] = (__bf16)a.y; o[2] = (__bf16)a.z; o[3] = (__bf16)a.w;
  o[4] = (__bf16)b.x; o[5] = (__bf16)b.y; o[6] = (__bf16)b.z; o[7] = (__bf16)b.w;
  *(bf16x8*)(y + i) = o;
}

// ---------------------------------------------------------------------------
// GEMM: Y[M,N] = (X[M,K] @ W[N,K]^T + bias[N]) * out_scale
// bf16 operands, f32 accumulate. Block tile 128x128, K staged 64 deep,
// double-buffered via async global->LDS copies (16 WMMAs per barrier round).
// 8 waves = 4(M) x 2(N); each wave owns 32x64 (2x4 frags).
// ---------------------------------------------------------------------------
template <bool OUT_BF16>
__global__ __launch_bounds__(256) void gemm_bias_bf16_kernel(
    const __bf16* __restrict__ X, const __bf16* __restrict__ W,
    const float* __restrict__ bias, void* __restrict__ Yv,
    int M, int N, int K, float out_scale) {
  __shared__ __attribute__((aligned(128))) __bf16 As[2][128][64];
  __shared__ __attribute__((aligned(128))) __bf16 Bs[2][128][64];

  const int tid  = threadIdx.x;
  const int lane = tid & 31;
  const int wave = tid >> 5;
  const int m0 = blockIdx.y * 128;
  const int n0 = blockIdx.x * 128;
  const int mw = (wave >> 1) * 32;
  const int nw = (wave & 1) * 64;

  f32x8 acc[2][4];
#pragma unroll
  for (int a = 0; a < 2; ++a)
#pragma unroll
    for (int b = 0; b < 4; ++b) acc[a][b] = zero_f32x8();

  // Staging map: thread t copies 64 bytes (4 x b128) of row (t>>1),
  // cols (t&1)*32.., for both the A (X) and B (W) tiles.
  const int srow = tid >> 1;
  const int scol = (tid & 1) * 32;

  auto issue_tile = [&](int buf, int k0) {
    const __bf16* gA = X + (size_t)(m0 + srow) * K + k0 + scol;
    const uint32_t lA = lds_off(&As[buf][srow][scol]);
#pragma unroll
    for (int i = 0; i < 4; ++i) async_copy_b128(lA + 16 * i, gA + 8 * i);
    const __bf16* gB = W + (size_t)(n0 + srow) * K + k0 + scol;
    const uint32_t lB = lds_off(&Bs[buf][srow][scol]);
#pragma unroll
    for (int i = 0; i < 4; ++i) async_copy_b128(lB + 16 * i, gB + 8 * i);
  };

  issue_tile(0, 0);
  const int NT = K / 64;
  for (int t = 0; t < NT; ++t) {
    wait_async0();       // this wave's copies of tile t landed
    __syncthreads();     // everyone's tile t landed; buffer (t+1)&1 free
    if (t + 1 < NT) issue_tile((t + 1) & 1, (t + 1) * 64);

    const __bf16(*A)[64] = As[t & 1];
    const __bf16(*B)[64] = Bs[t & 1];
#pragma unroll
    for (int ks = 0; ks < 64; ks += 32) {
      bf16x16 aF[2], bF[4];
#pragma unroll
      for (int a = 0; a < 2; ++a)
        aF[a] = wmma_a_frag(&A[mw + a * 16 + (lane & 15)][0], ks, lane);
#pragma unroll
      for (int b = 0; b < 4; ++b)
        bF[b] = wmma_b_frag(&B[nw + b * 16 + (lane & 15)][0], ks, lane);
#pragma unroll
      for (int a = 0; a < 2; ++a)
#pragma unroll
        for (int b = 0; b < 4; ++b)
          acc[a][b] = wmma_bf16(aF[a], bF[b], acc[a][b]);
    }
  }

  // C/D layout: lane = 16*khalf + n; VGPR v -> row (v + 8*khalf).
  const int col = lane & 15;
  const int rbase = (lane >> 4) * 8;
#pragma unroll
  for (int a = 0; a < 2; ++a)
#pragma unroll
    for (int b = 0; b < 4; ++b) {
      const int gn = n0 + nw + b * 16 + col;
      const float bv = bias[gn];
#pragma unroll
      for (int v = 0; v < 8; ++v) {
        const int gm = m0 + mw + a * 16 + rbase + v;
        const float r = (acc[a][b][v] + bv) * out_scale;
        if constexpr (OUT_BF16)
          ((__bf16*)Yv)[(size_t)gm * N + gn] = (__bf16)r;
        else
          // Final output: written once, never re-read -> keep it out of L2.
          __builtin_nontemporal_store(r, &((float*)Yv)[(size_t)gm * N + gn]);
      }
    }
}

// ---------------------------------------------------------------------------
// Flash attention over bf16 Q/K/V (Q pre-scaled by 1/sqrt(64) in its GEMM).
// Block = (b, h, 128-query tile); 8 waves, each owns 16 query rows.
// Q and K tiles staged via async copies; V transposed manually into LDS.
// ---------------------------------------------------------------------------
__global__ __launch_bounds__(256) void flash_attn_kernel(
    const __bf16* __restrict__ Qp, const __bf16* __restrict__ Kp,
    const __bf16* __restrict__ Vp, const int* __restrict__ mask,
    __bf16* __restrict__ Ctx) {
  __shared__ __attribute__((aligned(128))) __bf16 Qs[128][HEAD_DIM];
  __shared__ __attribute__((aligned(128))) __bf16 Ks[64][HEAD_DIM];   // [key][d]
  __shared__ __attribute__((aligned(128))) __bf16 Vts[HEAD_DIM][64];  // [d][key]
  __shared__ __attribute__((aligned(128))) __bf16 Pbuf[8][16][64];
  __shared__ float Sbuf[8][16][64];
  __shared__ float Corr[8][16];
  __shared__ float Lrow[8][16];

  const int tid  = threadIdx.x;
  const int lane = tid & 31;
  const int wave = tid >> 5;
  const int b  = blockIdx.z;
  const int h  = blockIdx.y;
  const int q0 = blockIdx.x * 128;

  const size_t headOff = (size_t)b * SEQ * D_MODEL + (size_t)h * HEAD_DIM;
  const __bf16* Qb = Qp + headOff;
  const __bf16* Kb = Kp + headOff;
  const __bf16* Vb = Vp + headOff;

  // Async-stage Q tile (pure byte copy; scale already folded in).
  {
    const int row = tid >> 1, c0 = (tid & 1) * 32;
    const __bf16* gq = Qb + (size_t)(q0 + row) * D_MODEL + c0;
    const uint32_t lq = lds_off(&Qs[row][c0]);
#pragma unroll
    for (int i = 0; i < 4; ++i) async_copy_b128(lq + 16 * i, gq + 8 * i);
  }
  wait_async0();
  __syncthreads();

  bf16x16 qF[2];
  qF[0] = wmma_a_frag(&Qs[wave * 16 + (lane & 15)][0], 0, lane);
  qF[1] = wmma_a_frag(&Qs[wave * 16 + (lane & 15)][0], 32, lane);

  f32x8 oF[4];
#pragma unroll
  for (int n = 0; n < 4; ++n) oF[n] = zero_f32x8();

  const int srow = lane >> 1;     // softmax: 2 lanes per query row
  const int shalf = lane & 1;
  const int cbase = shalf * 32;
  float m_i = -3.0e38f;
  float l_i = 0.0f;

  const int ccol = lane & 15;
  const int rbase = (lane >> 4) * 8;

  for (int c = 0; c < SEQ; c += 64) {
    __syncthreads();  // previous chunk's readers of Ks/Vts are done
    {
      const int row = tid >> 2, c0 = (tid & 3) * 16;
      // K chunk: async byte copy (row-major == B^T layout for QK^T).
      const __bf16* gk = Kb + (size_t)(c + row) * D_MODEL + c0;
      const uint32_t lk = lds_off(&Ks[row][c0]);
      async_copy_b128(lk, gk);
      async_copy_b128(lk + 16, gk + 8);
      // V chunk: load bf16 rows, scatter-transpose into Vts[d][key].
      const __bf16* gv = Vb + (size_t)(c + row) * D_MODEL + c0;
      bf16x8 v0 = *(const bf16x8*)gv;
      bf16x8 v1 = *(const bf16x8*)(gv + 8);
#pragma unroll
      for (int i = 0; i < 8; ++i) {
        Vts[c0 + i][row] = v0[i];
        Vts[c0 + 8 + i][row] = v1[i];
      }
    }
    wait_async0();
    __syncthreads();

    // S = Q K^T : 4 key n-tiles x 2 d-steps.
    f32x8 sF[4];
#pragma unroll
    for (int n = 0; n < 4; ++n) {
      const __bf16* kr = &Ks[n * 16 + (lane & 15)][0];
      bf16x16 b0 = wmma_b_frag(kr, 0, lane);
      bf16x16 b1 = wmma_b_frag(kr, 32, lane);
      f32x8 cc = zero_f32x8();
      cc = wmma_bf16(qF[0], b0, cc);
      cc = wmma_bf16(qF[1], b1, cc);
      sF[n] = cc;
    }
#pragma unroll
    for (int n = 0; n < 4; ++n)
#pragma unroll
      for (int v = 0; v < 8; ++v)
        Sbuf[wave][rbase + v][n * 16 + ccol] = sF[n][v];
    __syncthreads();

    // Online softmax: 2 lanes per row, 32 keys each, combine via shfl_xor(1).
    {
      const int* mrow = mask + ((size_t)b * SEQ + (q0 + wave * 16 + srow)) * SEQ + c + cbase;
      float mx = -3.0e38f;
      float sv[32];
#pragma unroll
      for (int jj = 0; jj < 32; jj += 4) {
        // Mask is streamed exactly once: non-temporal, keep L2 for K/V reuse.
        const i32x4 mv = __builtin_nontemporal_load((const i32x4*)(mrow + jj));
        float s0 = Sbuf[wave][srow][cbase + jj + 0];
        float s1 = Sbuf[wave][srow][cbase + jj + 1];
        float s2 = Sbuf[wave][srow][cbase + jj + 2];
        float s3 = Sbuf[wave][srow][cbase + jj + 3];
        s0 = (mv[0] == 0) ? -1.0e9f : s0;
        s1 = (mv[1] == 0) ? -1.0e9f : s1;
        s2 = (mv[2] == 0) ? -1.0e9f : s2;
        s3 = (mv[3] == 0) ? -1.0e9f : s3;
        sv[jj + 0] = s0; sv[jj + 1] = s1; sv[jj + 2] = s2; sv[jj + 3] = s3;
        mx = fmaxf(mx, fmaxf(fmaxf(s0, s1), fmaxf(s2, s3)));
      }
      mx = fmaxf(mx, __shfl_xor(mx, 1, 32));
      const float newm = fmaxf(m_i, mx);
      const float corr = __expf(m_i - newm);
      float lsum = 0.0f;
#pragma unroll
      for (int j = 0; j < 32; ++j) {
        const float p = __expf(sv[j] - newm);
        lsum += p;
        Pbuf[wave][srow][cbase + j] = (__bf16)p;
      }
      lsum += __shfl_xor(lsum, 1, 32);
      m_i = newm;
      l_i = l_i * corr + lsum;
      if (shalf == 0) Corr[wave][srow] = corr;
    }
    __syncthreads();

    // Rescale O by per-row correction, then O += P @ V.
    float cr[8];
#pragma unroll
    for (int v = 0; v < 8; ++v) cr[v] = Corr[wave][rbase + v];
#pragma unroll
    for (int n = 0; n < 4; ++n)
#pragma unroll
      for (int v = 0; v < 8; ++v) oF[n][v] *= cr[v];

    bf16x16 pF0 = wmma_a_frag(&Pbuf[wave][lane & 15][0], 0, lane);
    bf16x16 pF1 = wmma_a_frag(&Pbuf[wave][lane & 15][0], 32, lane);
#pragma unroll
    for (int n = 0; n < 4; ++n) {
      const __bf16* vr = &Vts[n * 16 + (lane & 15)][0];
      bf16x16 vb0 = wmma_b_frag(vr, 0, lane);
      bf16x16 vb1 = wmma_b_frag(vr, 32, lane);
      oF[n] = wmma_bf16(pF0, vb0, oF[n]);
      oF[n] = wmma_bf16(pF1, vb1, oF[n]);
    }
  }

  if (shalf == 0) Lrow[wave][srow] = l_i;
  __syncthreads();
  float inv[8];
#pragma unroll
  for (int v = 0; v < 8; ++v) inv[v] = 1.0f / Lrow[wave][rbase + v];
#pragma unroll
  for (int n = 0; n < 4; ++n)
#pragma unroll
    for (int v = 0; v < 8; ++v) {
      const int qrow = q0 + wave * 16 + rbase + v;
      const int d = n * 16 + ccol;
      Ctx[((size_t)b * SEQ + qrow) * D_MODEL + h * HEAD_DIM + d] =
          (__bf16)(oF[n][v] * inv[v]);
    }
}

extern "C" void kernel_launch(void* const* d_in, const int* in_sizes, int n_in,
                              void* d_out, int out_size, void* d_ws, size_t ws_size,
                              hipStream_t stream) {
  const float* query  = (const float*)d_in[0];
  const float* key_in = (const float*)d_in[1];
  const float* value  = (const float*)d_in[2];
  const int*   mask   = (const int*)d_in[3];
  const float* wq_w = (const float*)d_in[4];
  const float* wq_b = (const float*)d_in[5];
  const float* wk_w = (const float*)d_in[6];
  const float* wk_b = (const float*)d_in[7];
  const float* wv_w = (const float*)d_in[8];
  const float* wv_b = (const float*)d_in[9];
  const float* fc_w = (const float*)d_in[10];
  const float* fc_b = (const float*)d_in[11];
  float* out = (float*)d_out;

  const size_t ACT = (size_t)MROWS * D_MODEL;    // 8M elements
  const size_t WEL = (size_t)D_MODEL * D_MODEL;  // 1M elements
  __bf16* xq  = (__bf16*)d_ws;
  __bf16* xk  = xq + ACT;
  __bf16* xv  = xk + ACT;
  __bf16* wqb = xv + ACT;
  __bf16* wkb = wqb + WEL;
  __bf16* wvb = wkb + WEL;
  __bf16* wfb = wvb + WEL;
  __bf16* qp  = wfb + WEL;
  __bf16* kp  = qp + ACT;
  __bf16* vp  = kp + ACT;
  __bf16* ctx = vp + ACT;   // total 60M bf16 = 120 MB

  dim3 blk(256);
  // One-time bf16 conversions.
  cvt_f32_bf16_kernel<<<ACT / 2048, blk, 0, stream>>>(query,  xq);
  cvt_f32_bf16_kernel<<<ACT / 2048, blk, 0, stream>>>(key_in, xk);
  cvt_f32_bf16_kernel<<<ACT / 2048, blk, 0, stream>>>(value,  xv);
  cvt_f32_bf16_kernel<<<WEL / 2048, blk, 0, stream>>>(wq_w, wqb);
  cvt_f32_bf16_kernel<<<WEL / 2048, blk, 0, stream>>>(wk_w, wkb);
  cvt_f32_bf16_kernel<<<WEL / 2048, blk, 0, stream>>>(wv_w, wvb);
  cvt_f32_bf16_kernel<<<WEL / 2048, blk, 0, stream>>>(fc_w, wfb);

  dim3 gemmGrid(D_MODEL / 128, MROWS / 128);  // (8, 64)
  // Q projection pre-scaled by 1/sqrt(HEAD_DIM).
  gemm_bias_bf16_kernel<true><<<gemmGrid, blk, 0, stream>>>(
      xq, wqb, wq_b, qp, MROWS, D_MODEL, D_MODEL, 0.125f);
  gemm_bias_bf16_kernel<true><<<gemmGrid, blk, 0, stream>>>(
      xk, wkb, wk_b, kp, MROWS, D_MODEL, D_MODEL, 1.0f);
  gemm_bias_bf16_kernel<true><<<gemmGrid, blk, 0, stream>>>(
      xv, wvb, wv_b, vp, MROWS, D_MODEL, D_MODEL, 1.0f);

  dim3 attnGrid(SEQ / 128, N_HEADS, BSZ);     // (16, 16, 4)
  flash_attn_kernel<<<attnGrid, blk, 0, stream>>>(qp, kp, vp, mask, ctx);

  gemm_bias_bf16_kernel<false><<<gemmGrid, blk, 0, stream>>>(
      ctx, wfb, fc_b, out, MROWS, D_MODEL, D_MODEL, 1.0f);
}
